// ImplicitLatentAutoEncoder_7078106104295
// MI455X (gfx1250) — compile-verified, matrix operand order checked by reading, not run on hardware
//
#include <hip/hip_runtime.h>

typedef __attribute__((ext_vector_type(16))) _Float16 v16h;
typedef __attribute__((ext_vector_type(8)))  float    v8f;

#define EPSBN 1e-5f
#define Bn  64
#define Tn  32
#define On  32
#define En  16
#define Hn  128
#define Pn  992   /* O*(O-1) */
#define Om1 31

// ---------------------------------------------------------------------------
// GEMM: C[M,N] = relu_bn?(A[M,K]) @ Wt^T + bias, f16 WMMA, f32 accumulate.
// Wt is stored [N][K] f16. Optional per-K input BN (scale/shift) + ReLU fused
// into the A-tile load. Optional per-N column sum / sum^2 accumulation for
// training-mode BatchNorm of THIS layer's output. store_mode 1 scatters the
// final [2048,512] result to d_out as [B,T,O,E].
// Requirements: M % 64 == 0, K % 32 == 0 (all call sites satisfy this).
// ---------------------------------------------------------------------------
__global__ __launch_bounds__(256)
void gemm_f16_wmma(const float* __restrict__ A, int M, int K, int N,
                   const _Float16* __restrict__ Wt,
                   const float* __restrict__ bias,
                   const float* __restrict__ in_scale,
                   const float* __restrict__ in_shift,
                   float* __restrict__ C,
                   float* __restrict__ col_sum, float* __restrict__ col_sq,
                   int store_mode)
{
    __shared__ _Float16 As[64][40];    // 64 rows x 32 K (80B row stride, 16B aligned)
    __shared__ _Float16 Bs[128][40];   // 128 n-rows x 32 K
    __shared__ float s_sum[128], s_sq[128];

    const int tid  = threadIdx.x;
    const int lane = tid & 31;
    const int wid  = tid >> 5;
    const int wm   = wid & 3;                // 16-row strip within 64
    const int wn   = wid >> 2;               // 64-col strip within 128
    const int m0   = blockIdx.x * 64;
    const int n0   = blockIdx.y * 128;

    if (tid < 128) { s_sum[tid] = 0.f; s_sq[tid] = 0.f; }

    v8f acc[4] = {};
    const bool do_bn = (in_scale != nullptr);

    // staging indices (constant across K loop)
    const int ar = tid >> 2;                 // A row 0..63
    const int akb = (tid & 3) << 3;          // A k-base 0,8,16,24
    const int bn = tid >> 1;                 // B n-row 0..127
    const int bkb = (tid & 1) << 4;          // B k-base 0,16
    const int bgn = n0 + bn;

    for (int k0 = 0; k0 < K; k0 += 32) {
        __syncthreads();
        // ---- A tile: 64x32, two b128 loads -> (bn+relu) -> one b128 LDS store
        {
            const float* srcp = A + (size_t)(m0 + ar) * K + k0 + akb;
            float vv[8];
            *(float4*)&vv[0] = *(const float4*)(srcp);
            *(float4*)&vv[4] = *(const float4*)(srcp + 4);
            if (do_bn) {
                float ss[8], hh[8];
                *(float4*)&ss[0] = *(const float4*)(in_scale + k0 + akb);
                *(float4*)&ss[4] = *(const float4*)(in_scale + k0 + akb + 4);
                *(float4*)&hh[0] = *(const float4*)(in_shift + k0 + akb);
                *(float4*)&hh[4] = *(const float4*)(in_shift + k0 + akb + 4);
                #pragma unroll
                for (int j = 0; j < 8; ++j) {
                    float t = vv[j] * ss[j] + hh[j];
                    vv[j] = t > 0.f ? t : 0.f;
                }
            }
            union { _Float16 h[8]; uint4 u; } pk;
            #pragma unroll
            for (int j = 0; j < 8; ++j) pk.h[j] = (_Float16)vv[j];
            *(uint4*)&As[ar][akb] = pk.u;
        }
        // ---- B tile: 128x32 f16, two b128 copies per thread ----
        {
            uint4 u0 = make_uint4(0u, 0u, 0u, 0u);
            uint4 u1 = make_uint4(0u, 0u, 0u, 0u);
            if (bgn < N) {
                const uint4* srcp = (const uint4*)(Wt + (size_t)bgn * K + k0 + bkb);
                u0 = srcp[0];
                u1 = srcp[1];
            }
            uint4* d = (uint4*)&Bs[bn][bkb];
            d[0] = u0;
            d[1] = u1;
        }
        __syncthreads();

        // ---- fragments per ISA 16-bit A/B layout (lane[4] selects K half) --
        const int koff = ((lane >> 4) & 1) << 3;
        v16h afrag;
        {
            const _Float16* arp = &As[wm * 16 + (lane & 15)][0];
            #pragma unroll
            for (int v = 0; v < 8; ++v) {
                int k = ((v >> 2) << 4) + koff + ((v & 3) << 1);
                afrag[2 * v]     = arp[k];
                afrag[2 * v + 1] = arp[k + 1];
            }
        }
        #pragma unroll
        for (int nt = 0; nt < 4; ++nt) {
            v16h bfrag;
            const _Float16* brp = &Bs[wn * 64 + nt * 16 + (lane & 15)][0];
            #pragma unroll
            for (int v = 0; v < 8; ++v) {
                int k = ((v >> 2) << 4) + koff + ((v & 3) << 1);
                bfrag[2 * v]     = brp[k];
                bfrag[2 * v + 1] = brp[k + 1];
            }
            acc[nt] = __builtin_amdgcn_wmma_f32_16x16x32_f16(
                false, afrag, false, bfrag, (short)0, acc[nt], false, false);
        }
    }
    __syncthreads();

    // ---- epilogue: bias, store, BN column stats ----
    const bool do_stats = (col_sum != nullptr);
    #pragma unroll
    for (int nt = 0; nt < 4; ++nt) {
        int nl = wn * 64 + nt * 16 + (lane & 15);
        int gn = n0 + nl;
        if (gn >= N) continue;
        float bval = bias ? bias[gn] : 0.f;
        float psum = 0.f, psq = 0.f;
        #pragma unroll
        for (int r = 0; r < 8; ++r) {
            int ml = wm * 16 + r + (((lane >> 4) & 1) << 3);
            int gm = m0 + ml;
            float v = acc[nt][r] + bval;
            if (store_mode == 0) {
                C[(size_t)gm * N + gn] = v;
            } else { // [2048,512] -> [B,T,O,E]
                int b = gm >> 5, o = gm & 31, t = gn >> 4, e = gn & 15;
                C[((((size_t)b * Tn + t) * On + o) * En) + e] = v;
            }
            psum += v; psq += v * v;
        }
        if (do_stats) {
            atomicAdd(&s_sum[nl], psum);
            atomicAdd(&s_sq[nl],  psq);
        }
    }
    if (do_stats) {
        __syncthreads();
        if (tid < 128) {
            int gn = n0 + tid;
            if (gn < N) {
                atomicAdd(&col_sum[gn], s_sum[tid]);
                atomicAdd(&col_sq[gn],  s_sq[tid]);
            }
        }
    }
}

// x [B,T,O,E] -> source [B*O, T*E]
__global__ void k_transpose_x(const float* __restrict__ x, float* __restrict__ out)
{
    int idx = blockIdx.x * blockDim.x + threadIdx.x;
    if (idx >= 2048 * 512) return;
    int r = idx >> 9, c = idx & 511;
    int b = r >> 5, o = r & 31, t = c >> 4, e = c & 15;
    out[idx] = x[((((size_t)b * Tn + t) * On + o) * En) + e];
}

// W f32 [K][N] -> Wt f16 [N][K]
__global__ void k_wt_convert(const float* __restrict__ W, _Float16* __restrict__ Wt,
                             int K, int N)
{
    int idx = blockIdx.x * blockDim.x + threadIdx.x;
    if (idx >= K * N) return;
    int n = idx / K, k = idx - n * K;
    Wt[idx] = (_Float16)W[(size_t)k * N + n];
}

__global__ void k_zero(float* p, int n)
{
    int i = blockIdx.x * blockDim.x + threadIdx.x;
    if (i < n) p[i] = 0.f;
}

// edge_pre[b*P+p][c] = Pa[b,ii(p)][c] + Pb[b,jj(p)][c] + bias[c]  (+ BN stats)
__global__ __launch_bounds__(256)
void k_gather_add(const float* __restrict__ Pa, const float* __restrict__ Pb,
                  const float* __restrict__ bias, float* __restrict__ out,
                  float* __restrict__ col_sum, float* __restrict__ col_sq)
{
    __shared__ float s_s[256], s_q[256];
    int c    = threadIdx.x & 127;
    int half = threadIdx.x >> 7;
    int r0   = blockIdx.x * 16;
    float bv = bias[c];
    float sum = 0.f, sq = 0.f;
    for (int r = half; r < 16; r += 2) {
        int R = r0 + r;
        int b = R / Pn;
        int p = R - b * Pn;
        int i = p / Om1;
        int rem = p - i * Om1;
        int j = rem < i ? rem : rem + 1;
        float v = Pa[((size_t)b * On + i) * Hn + c]
                + Pb[((size_t)b * On + j) * Hn + c] + bv;
        out[(size_t)R * Hn + c] = v;
        sum += v; sq += v * v;
    }
    s_s[threadIdx.x] = sum; s_q[threadIdx.x] = sq;
    __syncthreads();
    if (threadIdx.x < 128) {
        atomicAdd(&col_sum[c], s_s[threadIdx.x] + s_s[threadIdx.x + 128]);
        atomicAdd(&col_sq[c],  s_q[threadIdx.x] + s_q[threadIdx.x + 128]);
    }
}

// mean over the 31 edges belonging to node (b,i)
__global__ void k_edge_mean(const float* __restrict__ edge, float* __restrict__ out)
{
    int c = threadIdx.x;      // 0..127
    int blk = blockIdx.x;     // b*32+i
    int b = blk >> 5, i = blk & 31;
    size_t base = ((size_t)b * Pn + (size_t)i * Om1) * Hn + c;
    float s = 0.f;
    for (int e = 0; e < Om1; ++e) s += edge[base + (size_t)e * Hn];
    out[(size_t)blk * Hn + c] = s * (1.0f / 31.0f);
}

__global__ void k_concat(const float* __restrict__ A, int wa,
                         const float* __restrict__ Bm, int wb,
                         float* __restrict__ out, int rows)
{
    int w = wa + wb;
    int idx = blockIdx.x * blockDim.x + threadIdx.x;
    if (idx >= rows * w) return;
    int r = idx / w, c = idx - r * w;
    out[idx] = (c < wa) ? A[(size_t)r * wa + c] : Bm[(size_t)r * wb + (c - wa)];
}

// scale = g * rsqrt(var+eps), shift = b - mean*scale (biased variance)
__global__ void k_bn_finalize(const float* __restrict__ sum, const float* __restrict__ sq,
                              const float* __restrict__ g, const float* __restrict__ bb,
                              float* __restrict__ scale, float* __restrict__ shift,
                              float invM)
{
    int i = threadIdx.x; // 128
    float m  = sum[i] * invM;
    float v  = sq[i]  * invM - m * m;
    float sc = g[i] * rsqrtf(v + EPSBN);
    scale[i] = sc;
    shift[i] = bb[i] - m * sc;
}

// ---------------------------------------------------------------------------
extern "C" void kernel_launch(void* const* d_in, const int* in_sizes, int n_in,
                              void* d_out, int out_size, void* d_ws, size_t ws_size,
                              hipStream_t stream)
{
    (void)in_sizes; (void)n_in; (void)out_size; (void)ws_size;
    const float* x = (const float*)d_in[0];
    auto F = [&](int i) { return (const float*)d_in[i]; };
    // Insertion-order flattening of the params dicts (fc1{W,b}, bn1{g,b}, ...)
    const float *W1e=F(1),*b1e=F(2),*g1e=F(3),*bnb1e=F(4);
    const float *W2e=F(5),*b2e=F(6),*g2e=F(7),*bnb2e=F(8);
    const float *W3e=F(9),*b3e=F(10);
    const float *W1n=F(11),*b1n=F(12),*g1n=F(13),*bnb1n=F(14);
    const float *W2n=F(15),*b2n=F(16),*g2n=F(17),*bnb2n=F(18);
    const float *W3n=F(19),*b3n=F(20),*g3n=F(21),*bnb3n=F(22);
    const float *W4n=F(23),*b4n=F(24);
    const float *W1d=F(25),*b1d=F(26),*g1d=F(27),*bnb1d=F(28);
    const float *W2d=F(29),*b2d=F(30),*g2d=F(31),*bnb2d=F(32);
    const float *W3d=F(33),*b3d=F(34),*g3d=F(35),*bnb3d=F(36);
    const float *W4d=F(37),*b4d=F(38);
    const float *W1m=F(39),*b1m=F(40),*g1m=F(41),*bnb1m=F(42);
    const float *W2m=F(43),*b2m=F(44),*g2m=F(45),*bnb2m=F(46);
    const float *W3m=F(47),*b3m=F(48),*g3m=F(49),*bnb3m=F(50);
    const float *W4m=F(51),*b4m=F(52);

    // ---- workspace carve ----
    char* base = (char*)d_ws;
    size_t off = 0;
    auto alloc = [&](size_t bytes) -> char* {
        char* p = base + off;
        off += (bytes + 255) & ~(size_t)255;
        return p;
    };
    float* src   = (float*)alloc(2048ull * 512 * 4);
    float* bufE0 = (float*)alloc(63488ull * 128 * 4);
    float* bufE1 = (float*)alloc(63488ull * 128 * 4);
    float* Pa    = (float*)alloc(2048ull * 128 * 4);   // reused as Qa
    float* Pb    = (float*)alloc(2048ull * 128 * 4);   // reused as Qb
    float* EM    = (float*)alloc(2048ull * 128 * 4);   // reused as DM
    float* CN    = (float*)alloc(2048ull * 640 * 4);   // reused as CD
    float* sA    = (float*)alloc(2048ull * 128 * 4);
    float* sB    = (float*)alloc(2048ull * 128 * 4);
    float* z     = (float*)alloc(2048ull * 32 * 4);
    float* stats = (float*)alloc(11ull * 512 * 4);     // [sum|sq|scale|shift] x 11
    _Float16* wt = (_Float16*)alloc(917504ull * 2);

    size_t wo = 0;
    auto walloc = [&](size_t halfs) { _Float16* p = wt + wo; wo += halfs; return p; };
    _Float16 *tW1ea=walloc(65536), *tW1eb=walloc(65536), *tW2e=walloc(16384), *tW3e=walloc(16384);
    _Float16 *tW1n=walloc(81920), *tW2n=walloc(16384), *tW3n=walloc(16384), *tW4n=walloc(4096);
    _Float16 *tW1da=walloc(4096), *tW1db=walloc(4096), *tW2d=walloc(16384), *tW3d=walloc(16384), *tW4d=walloc(16384);
    _Float16 *tW1m=walloc(20480), *tW2m=walloc(16384), *tW3m=walloc(16384), *tW4m=walloc(65536);

    auto conv = [&](const float* W, _Float16* dst, int K, int N) {
        int n = K * N;
        k_wt_convert<<<(n + 255) / 256, 256, 0, stream>>>(W, dst, K, N);
    };
    conv(W1e, tW1ea, 512, 128); conv(W1e + 512 * 128, tW1eb, 512, 128);
    conv(W2e, tW2e, 128, 128);  conv(W3e, tW3e, 128, 128);
    conv(W1n, tW1n, 640, 128);  conv(W2n, tW2n, 128, 128);
    conv(W3n, tW3n, 128, 128);  conv(W4n, tW4n, 128, 32);
    conv(W1d, tW1da, 32, 128);  conv(W1d + 32 * 128, tW1db, 32, 128);
    conv(W2d, tW2d, 128, 128);  conv(W3d, tW3d, 128, 128); conv(W4d, tW4d, 128, 128);
    conv(W1m, tW1m, 160, 128);  conv(W2m, tW2m, 128, 128);
    conv(W3m, tW3m, 128, 128);  conv(W4m, tW4m, 128, 512);

    k_zero<<<(11 * 512 + 255) / 256, 256, 0, stream>>>(stats, 11 * 512);
    k_transpose_x<<<(2048 * 512) / 256, 256, 0, stream>>>(x, src);

    auto gemm = [&](const float* A, int M, int K, int N, const _Float16* Wt,
                    const float* bias, int bn_in, int stat_out, float* C, int mode) {
        const float *sc = nullptr, *sh = nullptr;
        if (bn_in >= 0) { sc = stats + bn_in * 512 + 256; sh = stats + bn_in * 512 + 384; }
        float *cs = nullptr, *cq = nullptr;
        if (stat_out >= 0) { cs = stats + stat_out * 512; cq = stats + stat_out * 512 + 128; }
        dim3 grid(M / 64, (N + 127) / 128);
        gemm_f16_wmma<<<grid, 256, 0, stream>>>(A, M, K, N, Wt, bias, sc, sh, C, cs, cq, mode);
    };
    auto bnfin = [&](int L, const float* g, const float* bb, int count) {
        k_bn_finalize<<<1, 128, 0, stream>>>(stats + L * 512, stats + L * 512 + 128,
                                             g, bb, stats + L * 512 + 256,
                                             stats + L * 512 + 384, 1.0f / (float)count);
    };

    // ---- edge encoder (fc1 factorized through node projections) ----
    gemm(src, 2048, 512, 128, tW1ea, nullptr, -1, -1, Pa, 0);
    gemm(src, 2048, 512, 128, tW1eb, nullptr, -1, -1, Pb, 0);
    k_gather_add<<<63488 / 16, 256, 0, stream>>>(Pa, Pb, b1e, bufE0,
                                                 stats + 0 * 512, stats + 0 * 512 + 128);
    bnfin(0, g1e, bnb1e, 63488);
    gemm(bufE0, 63488, 128, 128, tW2e, b2e, 0, 1, bufE1, 0);
    bnfin(1, g2e, bnb2e, 63488);
    gemm(bufE1, 63488, 128, 128, tW3e, b3e, 1, -1, bufE0, 0);        // edge
    k_edge_mean<<<2048, 128, 0, stream>>>(bufE0, EM);

    // ---- node encoder ----
    k_concat<<<(2048 * 640 + 255) / 256, 256, 0, stream>>>(src, 512, EM, 128, CN, 2048);
    gemm(CN, 2048, 640, 128, tW1n, b1n, -1, 2, sA, 0);  bnfin(2, g1n, bnb1n, 2048);
    gemm(sA, 2048, 128, 128, tW2n, b2n, 2, 3, sB, 0);   bnfin(3, g2n, bnb2n, 2048);
    gemm(sB, 2048, 128, 128, tW3n, b3n, 3, 4, sA, 0);   bnfin(4, g3n, bnb3n, 2048);
    gemm(sA, 2048, 128, 32, tW4n, b4n, 4, -1, z, 0);                 // z [2048,32]

    // ---- edge decoder (fc1 factorized) ----
    gemm(z, 2048, 32, 128, tW1da, nullptr, -1, -1, Pa, 0);
    gemm(z, 2048, 32, 128, tW1db, nullptr, -1, -1, Pb, 0);
    k_gather_add<<<63488 / 16, 256, 0, stream>>>(Pa, Pb, b1d, bufE0,
                                                 stats + 5 * 512, stats + 5 * 512 + 128);
    bnfin(5, g1d, bnb1d, 63488);
    gemm(bufE0, 63488, 128, 128, tW2d, b2d, 5, 6, bufE1, 0); bnfin(6, g2d, bnb2d, 63488);
    gemm(bufE1, 63488, 128, 128, tW3d, b3d, 6, 7, bufE0, 0); bnfin(7, g3d, bnb3d, 63488);
    gemm(bufE0, 63488, 128, 128, tW4d, b4d, 7, -1, bufE1, 0);        // dedge
    k_edge_mean<<<2048, 128, 0, stream>>>(bufE1, EM);                 // dmean

    // ---- node decoder ----
    k_concat<<<(2048 * 160 + 255) / 256, 256, 0, stream>>>(z, 32, EM, 128, CN, 2048);
    gemm(CN, 2048, 160, 128, tW1m, b1m, -1, 8, sA, 0);  bnfin(8,  g1m, bnb1m, 2048);
    gemm(sA, 2048, 128, 128, tW2m, b2m, 8, 9, sB, 0);   bnfin(9,  g2m, bnb2m, 2048);
    gemm(sB, 2048, 128, 128, tW3m, b3m, 9, 10, sA, 0);  bnfin(10, g3m, bnb3m, 2048);
    gemm(sA, 2048, 128, 512, tW4m, b4m, 10, -1, (float*)d_out, 1);   // fused [B,T,O,E] store
}
